// RelationBiasedMultiHeadAttention_51573967290523
// MI455X (gfx1250) — matrix operand order repaired and checked by reading, not compiled, not run on hardware
//
#include <hip/hip_runtime.h>
#include <hip/hip_bf16.h>

// ---------------------------------------------------------------------------
// CDNA5 (gfx1250) relation-biased MHA, bf16 WMMA, flash-attention style.
// B=4, N=2048, HIDDEN=256, HEADS=8, HEAD_DIM=32.
// Round 3: fix async-to-LDS builtin pointer types (int4* generic pointers).
// ---------------------------------------------------------------------------

typedef __attribute__((ext_vector_type(16))) __bf16 v16bf;
typedef __attribute__((ext_vector_type(8)))  __bf16 v8bf;
typedef __attribute__((ext_vector_type(8)))  float  v8f;
typedef int async_v4i __attribute__((vector_size(16)));  // builtin's pointee type

#define HIDDEN   256
#define SEQ      2048
#define BATCH    4
#define HEADS    8
#define HDIM     32
#define ROWP     40   // padded LDS row stride (elements): 80B -> conflict-free

#if __has_builtin(__builtin_amdgcn_global_load_async_to_lds_b128) && \
    __has_builtin(__builtin_amdgcn_s_wait_asynccnt)
#define HAVE_ASYNC 1
// (global src, lds dst, offset, cpol) — pointers are generic int4*
#define GASYNC_CP16(g, l) __builtin_amdgcn_global_load_async_to_lds_b128( \
    (async_v4i*)(g), (async_v4i*)(l), 0, 0)
#else
#define HAVE_ASYNC 0
#endif

__device__ __forceinline__ v8f wmma_bf16(v16bf a, v16bf b, v8f c) {
    // D = A(16x32 bf16) x B(32x16 bf16) + C(16x16 f32)
    return __builtin_amdgcn_wmma_f32_16x16x32_bf16(
        false, a, false, b, (short)0, c, false, false);
}

// A-matrix (16x32, 16-bit) register layout used here (contiguous k, matching B; not ISA table order):
//   lane L: row M = L%16, half = L/16
//   element j: K = j + half*16
__device__ __forceinline__ v16bf loadA_bf16(const __bf16* row, int half) {
    v8bf r0 = *(const v8bf*)(row + half * 16);
    v8bf r1 = *(const v8bf*)(row + half * 16 + 8);
    v16bf r;
#pragma unroll
    for (int j = 0; j < 8; ++j) { r[j] = r0[j]; r[8 + j] = r1[j]; }
    return r;
}

__device__ __forceinline__ v16bf loadA_f32(const float* row, int half) {
    v16bf r;
#pragma unroll
    for (int j = 0; j < 8; ++j) {
        r[j]     = (__bf16)row[half * 16 + j];
        r[8 + j] = (__bf16)row[half * 16 + 8 + j];
    }
    return r;
}

// B-matrix (32x16, 16-bit): lane L: N = L%16, element j holds K = j + half*16
// -> 16 contiguous values; 16B-aligned pair load (LDS rows are 80B strided).
__device__ __forceinline__ v16bf loadB16(const __bf16* p) {
    v8bf a = *(const v8bf*)p;
    v8bf b = *(const v8bf*)(p + 8);
    v16bf r;
#pragma unroll
    for (int j = 0; j < 8; ++j) { r[j] = a[j]; r[8 + j] = b[j]; }
    return r;
}

__device__ __forceinline__ v16bf loadB_f32(const float* p) {
    v16bf r;
#pragma unroll
    for (int j = 0; j < 16; ++j) r[j] = (__bf16)p[j];
    return r;
}

// ---------------------------------------------------------------------------
// Kernel 1: QKV projections.  Y = X @ W^T + b, X=[8192,256] f32.
// blockIdx.y selects Q/K/V. Q,K -> [B,H,N,D] bf16 ; V -> [B,H,D,N] bf16.
// ---------------------------------------------------------------------------
__global__ __launch_bounds__(128) void qkv_proj_kernel(
    const float* __restrict__ hidden,
    const float* __restrict__ Wq, const float* __restrict__ bq,
    const float* __restrict__ Wk, const float* __restrict__ bk,
    const float* __restrict__ Wv, const float* __restrict__ bv,
    __bf16* __restrict__ qo, __bf16* __restrict__ ko, __bf16* __restrict__ vto)
{
    const int mode = blockIdx.y;                       // 0=Q 1=K 2=V
    const float* W    = (mode == 0) ? Wq : (mode == 1) ? Wk : Wv;
    const float* bias = (mode == 0) ? bq : (mode == 1) ? bk : bv;

    const int lane = threadIdx.x & 31;
    const int wid  = threadIdx.x >> 5;
    const int waveId = blockIdx.x * 4 + wid;           // 0..8191 tiles
    const int rbase = (waveId >> 4) * 16;
    const int cbase = (waveId & 15) * 16;
    const int half = lane >> 4, l16 = lane & 15;

    const float* arow = hidden + (size_t)(rbase + l16) * HIDDEN;
    const float* brow = W      + (size_t)(cbase + l16) * HIDDEN;

    v8f acc = {};
#pragma unroll
    for (int kc = 0; kc < HIDDEN / 32; ++kc) {
        v16bf a = loadA_f32(arow + kc * 32, half);
        v16bf b = loadB_f32(brow + kc * 32 + half * 16);  // B[k,n] = W[n,k]
        acc = wmma_bf16(a, b, acc);
    }

    const float bn = bias[cbase + l16];
    const int c = cbase + l16, h = c >> 5, d = c & 31;
#pragma unroll
    for (int r = 0; r < 8; ++r) {
        const int grow = rbase + half * 8 + r;          // b*2048 + n
        const int bb = grow >> 11, nn = grow & 2047;
        const float y = acc[r] + bn;
        if (mode == 2) {
            vto[(((size_t)bb * HEADS + h) * HDIM + d) * SEQ + nn] = (__bf16)y;
        } else {
            __bf16* dst = (mode == 0) ? qo : ko;
            dst[(((size_t)bb * HEADS + h) * SEQ + nn) * HDIM + d] = (__bf16)y;
        }
    }
}

// ---------------------------------------------------------------------------
// Kernel 2: flash attention with relation bias.
// Block = 4 waves, one (b,h), 4 query tiles; 32-key K/V chunks staged in LDS
// once per block, double-buffered (async-to-LDS when available).
// ---------------------------------------------------------------------------
__global__ __launch_bounds__(128) void attn_kernel(
    const __bf16* __restrict__ q,   // [B,H,N,D]
    const __bf16* __restrict__ k,   // [B,H,N,D]
    const __bf16* __restrict__ vt,  // [B,H,D,N]
    const float*  __restrict__ rel_bias,   // [B,N,N]
    const float*  __restrict__ head_scale, // [H]
    __bf16* __restrict__ attn_out)         // [B*N, HIDDEN]
{
    __shared__ __attribute__((aligned(32))) __bf16 kbuf[2][32 * ROWP];
    __shared__ __attribute__((aligned(32))) __bf16 vbuf[2][32 * ROWP];
    __shared__ __attribute__((aligned(32))) __bf16 pbuf[4][16 * ROWP];

    const int tid  = threadIdx.x;
    const int lane = tid & 31;
    const int wid  = tid >> 5;
    const int bh   = blockIdx.x >> 5;          // 0..31 = b*8 + h
    const int b = bh >> 3, h = bh & 7;
    const int qt = (blockIdx.x & 31) * 4 + wid;
    const int half = lane >> 4, l16 = lane & 15;
    const int qbase = qt * 16;

    const __bf16* qhead = q  + ((size_t)bh * SEQ) * HDIM;
    const __bf16* khead = k  + ((size_t)bh * SEQ) * HDIM;
    const __bf16* vhead = vt + ((size_t)bh * HDIM) * SEQ;
    const float* biasRow = rel_bias + (size_t)b * SEQ * SEQ + (size_t)qbase * SEQ;
    const float hs = head_scale[h];
    const float scale = 0.17677669529663687f;   // 1/sqrt(32)

    // Per-thread staging slot: row = tid/4 (key for K, d for V), 8-elem segment
    const int srow = tid >> 2, sseg = (tid & 3) * 8;

    // Q tile in A layout, loaded once.
    const v16bf qa = loadA_bf16(qhead + (size_t)(qbase + l16) * HDIM, half);

    v8f o0 = {}, o1 = {};
    float rmax[8], rsum[8];
#pragma unroll
    for (int r = 0; r < 8; ++r) { rmax[r] = -1e30f; rsum[r] = 0.0f; }

    __bf16* myP = pbuf[wid];

    // ---- prologue: stage chunk 0 into buffer 0 ----
#if HAVE_ASYNC
    GASYNC_CP16(khead + (size_t)srow * HDIM + sseg, &kbuf[0][srow * ROWP + sseg]);
    GASYNC_CP16(vhead + (size_t)srow * SEQ + sseg,  &vbuf[0][srow * ROWP + sseg]);
    __builtin_amdgcn_s_wait_asynccnt(0);
#else
    *(v8bf*)&kbuf[0][srow * ROWP + sseg] =
        *(const v8bf*)(khead + (size_t)srow * HDIM + sseg);
    *(v8bf*)&vbuf[0][srow * ROWP + sseg] =
        *(const v8bf*)(vhead + (size_t)srow * SEQ + sseg);
#endif
    __syncthreads();

    for (int it = 0; it < SEQ / 32; ++it) {
        const int nb = it * 32;
        const int cur = it & 1, nxt = cur ^ 1;
        const bool more = (it + 1) < (SEQ / 32);

        // ---- issue staging of next chunk (overlaps compute below) ----
#if HAVE_ASYNC
        if (more) {
            GASYNC_CP16(khead + (size_t)(nb + 32 + srow) * HDIM + sseg,
                        &kbuf[nxt][srow * ROWP + sseg]);
            GASYNC_CP16(vhead + (size_t)srow * SEQ + nb + 32 + sseg,
                        &vbuf[nxt][srow * ROWP + sseg]);
        }
#else
        v8bf kreg = {}, vreg = {};
        if (more) {
            kreg = *(const v8bf*)(khead + (size_t)(nb + 32 + srow) * HDIM + sseg);
            vreg = *(const v8bf*)(vhead + (size_t)srow * SEQ + nb + 32 + sseg);
        }
#endif

        const __bf16* kT = kbuf[cur];
        const __bf16* vT = vbuf[cur];

        // ---- S = Q @ K^T (two 16-key halves); B[d,n] = K[nb+n, d] ----
        v16bf kb0 = loadB16(kT + l16 * ROWP + half * 16);
        v16bf kb1 = loadB16(kT + (16 + l16) * ROWP + half * 16);
        v8f z = {};
        v8f s0 = wmma_bf16(qa, kb0, z);
        v8f s1 = wmma_bf16(qa, kb1, z);

        // ---- scale + relation bias; online softmax stats ----
        float ns0[8], ns1[8], alpha[8];
#pragma unroll
        for (int r = 0; r < 8; ++r) {
            const int m = half * 8 + r;
            const float* bp = biasRow + (size_t)m * SEQ + nb;
            if (nb + 64 < SEQ) __builtin_prefetch(bp + 64, 0, 3); // 2 chunks ahead
            ns0[r] = s0[r] * scale + bp[l16] * hs;
            ns1[r] = s1[r] * scale + bp[16 + l16] * hs;
            float cm = fmaxf(ns0[r], ns1[r]);
            cm = fmaxf(cm, __shfl_xor(cm, 1, 32));
            cm = fmaxf(cm, __shfl_xor(cm, 2, 32));
            cm = fmaxf(cm, __shfl_xor(cm, 4, 32));
            cm = fmaxf(cm, __shfl_xor(cm, 8, 32));
            const float nm = fmaxf(rmax[r], cm);
            alpha[r] = __expf(rmax[r] - nm);
            rmax[r] = nm;
        }

        // ---- P = exp(S - max) -> LDS (C layout); rescale O, L ----
#pragma unroll
        for (int r = 0; r < 8; ++r) {
            const int m = half * 8 + r;
            const float p0 = __expf(ns0[r] - rmax[r]);
            const float p1 = __expf(ns1[r] - rmax[r]);
            myP[m * ROWP + l16]      = (__bf16)p0;
            myP[m * ROWP + 16 + l16] = (__bf16)p1;
            float ps = p0 + p1;
            ps += __shfl_xor(ps, 1, 32);
            ps += __shfl_xor(ps, 2, 32);
            ps += __shfl_xor(ps, 4, 32);
            ps += __shfl_xor(ps, 8, 32);
            rsum[r] = rsum[r] * alpha[r] + ps;
            o0[r] = o0[r] * alpha[r];
            o1[r] = o1[r] * alpha[r];
        }

        asm volatile("" ::: "memory");   // order P store -> P reload (HW: per-wave LDS in-order)

        // ---- O += P @ V; P in A layout, V tile is [d][key] in LDS ----
        v16bf pa  = loadA_bf16(myP + l16 * ROWP, half);
        v16bf vb0 = loadB16(vT + l16 * ROWP + half * 16);
        v16bf vb1 = loadB16(vT + (16 + l16) * ROWP + half * 16);
        o0 = wmma_bf16(pa, vb0, o0);
        o1 = wmma_bf16(pa, vb1, o1);

        // ---- retire staging, flip buffers ----
#if HAVE_ASYNC
        __builtin_amdgcn_s_wait_asynccnt(0);
#else
        if (more) {
            *(v8bf*)&kbuf[nxt][srow * ROWP + sseg] = kreg;
            *(v8bf*)&vbuf[nxt][srow * ROWP + sseg] = vreg;
        }
#endif
        __syncthreads();
    }

    // ---- normalize and emit [B*N, HIDDEN] bf16 ----
#pragma unroll
    for (int r = 0; r < 8; ++r) {
        const int m = half * 8 + r;
        const float inv = 1.0f / rsum[r];
        const size_t grow = (size_t)b * SEQ + qbase + m;
        attn_out[grow * HIDDEN + h * HDIM + l16]      = (__bf16)(o0[r] * inv);
        attn_out[grow * HIDDEN + h * HDIM + 16 + l16] = (__bf16)(o1[r] * inv);
    }
}

// ---------------------------------------------------------------------------
// Kernel 3: output projection.  out = attn_out(bf16) @ Wo^T + bo  -> f32
// ---------------------------------------------------------------------------
__global__ __launch_bounds__(128) void out_proj_kernel(
    const __bf16* __restrict__ x,   // [8192, 256] bf16
    const float* __restrict__ Wo, const float* __restrict__ bo,
    float* __restrict__ out)        // [8192, 256] f32
{
    const int lane = threadIdx.x & 31;
    const int wid  = threadIdx.x >> 5;
    const int waveId = blockIdx.x * 4 + wid;
    const int rbase = (waveId >> 4) * 16;
    const int cbase = (waveId & 15) * 16;
    const int half = lane >> 4, l16 = lane & 15;

    const __bf16* arow = x  + (size_t)(rbase + l16) * HIDDEN;
    const float*  brow = Wo + (size_t)(cbase + l16) * HIDDEN;

    v8f acc = {};
#pragma unroll
    for (int kc = 0; kc < HIDDEN / 32; ++kc) {
        v16bf a = loadA_bf16(arow + kc * 32, half);
        v16bf b = loadB_f32(brow + kc * 32 + half * 16);
        acc = wmma_bf16(a, b, acc);
    }

    const float bn = bo[cbase + l16];
#pragma unroll
    for (int r = 0; r < 8; ++r)
        out[(size_t)(rbase + half * 8 + r) * HIDDEN + cbase + l16] = acc[r] + bn;
}

// ---------------------------------------------------------------------------
extern "C" void kernel_launch(void* const* d_in, const int* in_sizes, int n_in,
                              void* d_out, int out_size, void* d_ws, size_t ws_size,
                              hipStream_t stream) {
    const float* hidden     = (const float*)d_in[0];
    const float* rel_bias   = (const float*)d_in[1];
    const float* Wq = (const float*)d_in[2];
    const float* bq = (const float*)d_in[3];
    const float* Wk = (const float*)d_in[4];
    const float* bk = (const float*)d_in[5];
    const float* Wv = (const float*)d_in[6];
    const float* bv = (const float*)d_in[7];
    const float* Wo = (const float*)d_in[8];
    const float* bo = (const float*)d_in[9];
    const float* head_scale = (const float*)d_in[10];
    float* out = (float*)d_out;

    // ws layout: q (4MB) | k (4MB) | vt (4MB) | attn_out (4MB)  = 16MB bf16
    const size_t SEG = (size_t)BATCH * HEADS * SEQ * HDIM * sizeof(__bf16);
    __bf16* qws  = (__bf16*)((char*)d_ws + 0 * SEG);
    __bf16* kws  = (__bf16*)((char*)d_ws + 1 * SEG);
    __bf16* vws  = (__bf16*)((char*)d_ws + 2 * SEG);
    __bf16* aows = (__bf16*)((char*)d_ws + 3 * SEG);

    dim3 gProj(2048, 3, 1);
    qkv_proj_kernel<<<gProj, 128, 0, stream>>>(hidden, Wq, bq, Wk, bk, Wv, bv,
                                               qws, kws, vws);

    attn_kernel<<<1024, 128, 0, stream>>>(qws, kws, vws, rel_bias, head_scale, aows);

    out_proj_kernel<<<2048, 128, 0, stream>>>(aows, Wo, bo, out);
}
